// MyMultiAttentionLayer_25640954757808
// MI455X (gfx1250) — matrix operand order repaired and checked by reading, not compiled, not run on hardware
//
#include <hip/hip_runtime.h>

// ---------------------------------------------------------------------------
// Multi-head attention (softmax over the QUERY axis) for MI455X / gfx1250.
// bf16 WMMA (v_wmma_f32_16x16x32_bf16) for all GEMMs, fp32 accumulate.
// Two-pass column softmax: never materializes the [B,H,S,S] score tensor.
// 32x64 output tile per wave -> every B fragment feeds two WMMAs.
// ---------------------------------------------------------------------------

namespace {
constexpr int Bn  = 2;
constexpr int Sn  = 2048;
constexpr int Dn  = 1024;
constexpr int Hn  = 16;
constexpr int Pn  = 64;
constexpr int HPn = Hn * Pn;   // 1024
constexpr int NOn = 1024;
}

typedef __attribute__((ext_vector_type(16))) __bf16 v16bf;
typedef __attribute__((ext_vector_type(8)))  float  v8f;

union FragU { v16bf v; unsigned int u[8]; };

__device__ __forceinline__ unsigned short f2bf(float f) {
  unsigned int u = __builtin_bit_cast(unsigned int, f);
  u += 0x7FFFu + ((u >> 16) & 1u);          // round-to-nearest-even
  return (unsigned short)(u >> 16);
}

__device__ __forceinline__ v8f zero8() {
  v8f z = {0.f, 0.f, 0.f, 0.f, 0.f, 0.f, 0.f, 0.f};
  return z;
}

__device__ __forceinline__ v8f wmma_bf16(v16bf a, v16bf b, v8f c) {
  // (neg_a, A, neg_b, B, c_mod, C, reuse_a, reuse_b)
  return __builtin_amdgcn_wmma_f32_16x16x32_bf16(false, a, false, b,
                                                 (short)0, c, false, false);
}

// A fragment: 16x32 bf16, src row-major [M][K], row stride ld (elements).
// A layout (matches B): lanes0-15 -> K 0..15, lanes16-31 -> K 16..31.
__device__ __forceinline__ v16bf load_a_frag(const unsigned short* src, int ld, int lane) {
  FragU f;
  const int m  = lane & 15;
  const int h8 = (lane >> 4) << 4;          // 0 or 16
  const unsigned short* row = src + m * ld;
#pragma unroll
  for (int v = 0; v < 8; ++v) {
    const int k = h8 + (v << 1);
    f.u[v] = *reinterpret_cast<const unsigned int*>(row + k);
  }
  return f.v;
}

// B fragment for NT gemm: logical B[k][n] = src[n*ld + k] (src row-major [N][K]).
// B layout used here (same as A): lanes0-15 hold K=0..15 (V0..V7), lanes16-31 hold K=16..31.
__device__ __forceinline__ v16bf load_b_frag_nt(const unsigned short* src, int ld, int lane) {
  FragU f;
  const int n   = lane & 15;
  const int k16 = (lane >> 4) << 4;         // 0 or 16
  const unsigned short* row = src + n * ld + k16;
#pragma unroll
  for (int v = 0; v < 8; ++v)
    f.u[v] = *reinterpret_cast<const unsigned int*>(row + (v << 1));
  return f.v;
}

// ---------------------------------------------------------------------------
__global__ void convert_f32_bf16(const float* __restrict__ in,
                                 unsigned short* __restrict__ out, int n) {
  int i = blockIdx.x * blockDim.x + threadIdx.x;
  const int stride = gridDim.x * blockDim.x;
  for (; i < n; i += stride) out[i] = f2bf(in[i]);
}

// C[m,n] = sum_d A[m,d]*W[n,d] + bias[n];  A:[4096xD] W:[HPxD] bf16.
// One wave -> 32 rows x 64 cols (2 m-subtiles x 4 n-tiles; B frags reused 2x).
// transpose_out: 0 -> [B,H,S,P], 1 -> [B,H,P,S] (V, so attn x V is NT too).
__global__ void proj_kernel(const unsigned short* __restrict__ A,
                            const unsigned short* __restrict__ W,
                            const float* __restrict__ bias,
                            unsigned short* __restrict__ out, int transpose_out) {
  const int wave = (blockIdx.x * blockDim.x + threadIdx.x) >> 5;
  const int lane = threadIdx.x & 31;
  const int mt = wave >> 4;                 // 128 m-tiles of 32 rows
  const int ng = wave & 15;                 // 16 groups of 64 cols
  const int m0 = mt * 32;
  const int n0 = ng * 64;

  v8f acc[2][4];
#pragma unroll
  for (int i = 0; i < 2; ++i)
#pragma unroll
    for (int j = 0; j < 4; ++j) acc[i][j] = zero8();

  for (int k0 = 0; k0 < Dn; k0 += 32) {
    v16bf a0 = load_a_frag(A + (size_t)m0 * Dn + k0, Dn, lane);
    v16bf a1 = load_a_frag(A + (size_t)(m0 + 16) * Dn + k0, Dn, lane);
#pragma unroll
    for (int j = 0; j < 4; ++j) {
      v16bf b = load_b_frag_nt(W + (size_t)(n0 + 16 * j) * Dn + k0, Dn, lane);
      acc[0][j] = wmma_bf16(a0, b, acc[0][j]);
      acc[1][j] = wmma_bf16(a1, b, acc[1][j]);
    }
  }

  const int nlo   = lane & 15;
  const int rbase = (lane >> 4) << 3;       // 0 or 8
#pragma unroll
  for (int i = 0; i < 2; ++i) {
#pragma unroll
    for (int j = 0; j < 4; ++j) {
      const int n = n0 + 16 * j + nlo;
      const int hh = n >> 6;                // head
      const int p  = n & 63;
      const float bs = bias[n];
#pragma unroll
      for (int r = 0; r < 8; ++r) {
        const int m = m0 + 16 * i + rbase + r;
        const int b = m >> 11;              // / Sn
        const int s = m & (Sn - 1);
        const float val = acc[i][j][r] + bs;
        size_t idx;
        if (transpose_out) idx = ((size_t)((b * Hn + hh) * Pn + p)) * Sn + s;
        else               idx = ((size_t)((b * Hn + hh) * Sn + s)) * Pn + p;
        out[idx] = f2bf(val);
      }
    }
  }
}

// Pass 1 of column softmax: for each (b,h,k) compute max_q and sum_q exp of
// scores[q,k] = (Q[q,:] . K[k,:]) / 8. One wave handles 32 k-columns
// (2 k-tiles) so each streamed Q fragment feeds 2 WMMAs.
__global__ void colstats_kernel(const unsigned short* __restrict__ qb,
                                const unsigned short* __restrict__ kb,
                                float* __restrict__ colmax,
                                float* __restrict__ colsum) {
  const int wave = (blockIdx.x * blockDim.x + threadIdx.x) >> 5;
  const int lane = threadIdx.x & 31;
  const int kt2 = wave & 63;                // S/32 groups of 32 columns
  const int bh  = wave >> 6;
  const int kbase = kt2 * 32;
  const unsigned short* Q = qb + (size_t)bh * Sn * Pn;
  const unsigned short* K = kb + (size_t)bh * Sn * Pn;

  v16bf bk[2][2];
#pragma unroll
  for (int t = 0; t < 2; ++t) {
    bk[t][0] = load_b_frag_nt(K + (size_t)(kbase + 16 * t) * Pn +  0, Pn, lane);
    bk[t][1] = load_b_frag_nt(K + (size_t)(kbase + 16 * t) * Pn + 32, Pn, lane);
  }

  float mx[2] = {-3.0e38f, -3.0e38f};
  float sm[2] = {0.f, 0.f};
  for (int qt = 0; qt < Sn / 16; ++qt) {
    v16bf a0 = load_a_frag(Q + (size_t)qt * 16 * Pn +  0, Pn, lane);
    v16bf a1 = load_a_frag(Q + (size_t)qt * 16 * Pn + 32, Pn, lane);
#pragma unroll
    for (int t = 0; t < 2; ++t) {
      v8f s = zero8();
      s = wmma_bf16(a0, bk[t][0], s);
      s = wmma_bf16(a1, bk[t][1], s);
      float e[8];
      float tmax = -3.0e38f;
#pragma unroll
      for (int r = 0; r < 8; ++r) { e[r] = s[r] * 0.125f; tmax = fmaxf(tmax, e[r]); }
      const float nm = fmaxf(mx[t], tmax);
      float part = 0.f;
#pragma unroll
      for (int r = 0; r < 8; ++r) part += __expf(e[r] - nm);
      sm[t] = sm[t] * __expf(mx[t] - nm) + part;
      mx[t] = nm;
    }
  }
  // combine the two row-halves (lane n holds rows {0-7} or {8-15} of column n)
#pragma unroll
  for (int t = 0; t < 2; ++t) {
    const float omx = __shfl_xor(mx[t], 16, 32);
    const float osm = __shfl_xor(sm[t], 16, 32);
    const float M = fmaxf(mx[t], omx);
    const float T = sm[t] * __expf(mx[t] - M) + osm * __expf(omx - M);
    if (lane < 16) {
      const size_t k = (size_t)bh * Sn + kbase + 16 * t + lane;
      colmax[k] = M;
      colsum[k] = T;
    }
  }
}

// Pass 2: recompute score tiles (bitwise-identical WMMA sequence), apply
// exp((s-colmax))/colsum, transpose C-layout -> A-layout via LDS, then
// ctx[q,p] += attn[q,k] * V[k,p] with WMMA (Vt stored [B,H,P,S] -> NT gemm).
// One wave handles 32 q-rows (2 q-tiles) so K/V fragments are reused 2x.
__global__ void ctx_kernel(const unsigned short* __restrict__ qb,
                           const unsigned short* __restrict__ kb,
                           const unsigned short* __restrict__ vt,
                           const float* __restrict__ colmax,
                           const float* __restrict__ colsum,
                           unsigned short* __restrict__ ctxb) {
  __shared__ unsigned short lds_attn[4][2][16 * 32];  // 8 KB / block
  const int wid  = threadIdx.x >> 5;
  const int lane = threadIdx.x & 31;
  const int wave = blockIdx.x * 4 + wid;
  const int qt2 = wave & 63;                // S/32 groups of 32 q-rows
  const int bh  = wave >> 6;
  const int b = bh >> 4, h = bh & 15;
  const int q0 = qt2 * 32;

  const unsigned short* Q = qb + (size_t)bh * Sn * Pn;
  const unsigned short* K = kb + (size_t)bh * Sn * Pn;
  const unsigned short* V = vt + (size_t)bh * Pn * Sn;
  const float* cm = colmax + (size_t)bh * Sn;
  const float* cs = colsum + (size_t)bh * Sn;

  v16bf aq[2][2];
#pragma unroll
  for (int i = 0; i < 2; ++i) {
    aq[i][0] = load_a_frag(Q + (size_t)(q0 + 16 * i) * Pn +  0, Pn, lane);
    aq[i][1] = load_a_frag(Q + (size_t)(q0 + 16 * i) * Pn + 32, Pn, lane);
  }

  v8f acc[2][4];
#pragma unroll
  for (int i = 0; i < 2; ++i)
#pragma unroll
    for (int j = 0; j < 4; ++j) acc[i][j] = zero8();

  const int nlo   = lane & 15;
  const int rbase = (lane >> 4) << 3;

  for (int k0 = 0; k0 < Sn; k0 += 32) {
    // prefetch next 32x64 K chunk (4 KB): one 128B line per lane
    __builtin_prefetch(K + (size_t)(k0 + 32) * Pn + lane * 64, 0, 1);
#pragma unroll
    for (int t = 0; t < 2; ++t) {           // k-subtiles
      const int kt = k0 + 16 * t;
      const v16bf bkA = load_b_frag_nt(K + (size_t)kt * Pn +  0, Pn, lane);
      const v16bf bkB = load_b_frag_nt(K + (size_t)kt * Pn + 32, Pn, lane);
      const float m  = cm[kt + nlo];
      const float rs = 1.0f / cs[kt + nlo];
#pragma unroll
      for (int i = 0; i < 2; ++i) {         // q-subtiles
        v8f s = zero8();
        s = wmma_bf16(aq[i][0], bkA, s);
        s = wmma_bf16(aq[i][1], bkB, s);
#pragma unroll
        for (int r = 0; r < 8; ++r) {
          const float a = __expf(s[r] * 0.125f - m) * rs;
          lds_attn[wid][i][(rbase + r) * 32 + 16 * t + nlo] = f2bf(a);
        }
      }
    }
    asm volatile("s_wait_dscnt 0x0" ::: "memory");
    const v16bf af0 = load_a_frag(lds_attn[wid][0], 32, lane);
    const v16bf af1 = load_a_frag(lds_attn[wid][1], 32, lane);
#pragma unroll
    for (int j = 0; j < 4; ++j) {
      v16bf bv = load_b_frag_nt(V + (size_t)(16 * j) * Sn + k0, Sn, lane);
      acc[0][j] = wmma_bf16(af0, bv, acc[0][j]);
      acc[1][j] = wmma_bf16(af1, bv, acc[1][j]);
    }
    // per-wave LDS ops are in-order: next iteration's stores can't pass loads
  }

#pragma unroll
  for (int i = 0; i < 2; ++i) {
#pragma unroll
    for (int j = 0; j < 4; ++j) {
      const int p = 16 * j + nlo;
#pragma unroll
      for (int r = 0; r < 8; ++r) {
        const int q = q0 + 16 * i + rbase + r;
        ctxb[((size_t)(b * Sn + q)) * HPn + h * Pn + p] = f2bf(acc[i][j][r]);
      }
    }
  }
}

// out[m,n] = sum_c ctx[m,c] * LW[n,c] + lb[n]  (fp32 output)
__global__ void final_kernel(const unsigned short* __restrict__ ctxb,
                             const unsigned short* __restrict__ lw,
                             const float* __restrict__ lb,
                             float* __restrict__ out) {
  const int wave = (blockIdx.x * blockDim.x + threadIdx.x) >> 5;
  const int lane = threadIdx.x & 31;
  const int mt = wave >> 4;                 // 128 m-tiles of 32 rows
  const int ng = wave & 15;
  const int m0 = mt * 32;
  const int n0 = ng * 64;

  v8f acc[2][4];
#pragma unroll
  for (int i = 0; i < 2; ++i)
#pragma unroll
    for (int j = 0; j < 4; ++j) acc[i][j] = zero8();

  for (int k0 = 0; k0 < HPn; k0 += 32) {
    v16bf a0 = load_a_frag(ctxb + (size_t)m0 * HPn + k0, HPn, lane);
    v16bf a1 = load_a_frag(ctxb + (size_t)(m0 + 16) * HPn + k0, HPn, lane);
#pragma unroll
    for (int j = 0; j < 4; ++j) {
      v16bf bfr = load_b_frag_nt(lw + (size_t)(n0 + 16 * j) * HPn + k0, HPn, lane);
      acc[0][j] = wmma_bf16(a0, bfr, acc[0][j]);
      acc[1][j] = wmma_bf16(a1, bfr, acc[1][j]);
    }
  }

  const int nlo   = lane & 15;
  const int rbase = (lane >> 4) << 3;
#pragma unroll
  for (int i = 0; i < 2; ++i) {
#pragma unroll
    for (int j = 0; j < 4; ++j) {
      const int n = n0 + 16 * j + nlo;
      const float bs = lb[n];
#pragma unroll
      for (int r = 0; r < 8; ++r) {
        const int m = m0 + 16 * i + rbase + r;
        out[(size_t)m * NOn + n] = acc[i][j][r] + bs;
      }
    }
  }
}

// ---------------------------------------------------------------------------
extern "C" void kernel_launch(void* const* d_in, const int* in_sizes, int n_in,
                              void* d_out, int out_size, void* d_ws, size_t ws_size,
                              hipStream_t stream) {
  (void)in_sizes; (void)n_in; (void)out_size; (void)ws_size;
  const float* x    = (const float*)d_in[0];
  const float* q_w  = (const float*)d_in[1];
  const float* q_b  = (const float*)d_in[2];
  const float* k_w  = (const float*)d_in[3];
  const float* k_b  = (const float*)d_in[4];
  const float* v_w  = (const float*)d_in[5];
  const float* v_b  = (const float*)d_in[6];
  const float* l_w  = (const float*)d_in[7];
  const float* l_b  = (const float*)d_in[8];
  float* out = (float*)d_out;

  char* ws = (char*)d_ws;
  size_t off = 0;
  auto take_u16 = [&](size_t elems) {
    unsigned short* p = (unsigned short*)(ws + off);
    off = (off + elems * 2 + 255) & ~(size_t)255;
    return p;
  };
  auto take_f32 = [&](size_t elems) {
    float* p = (float*)(ws + off);
    off = (off + elems * 4 + 255) & ~(size_t)255;
    return p;
  };

  unsigned short* xb   = take_u16((size_t)Bn * Sn * Dn);
  unsigned short* wqb  = take_u16((size_t)HPn * Dn);
  unsigned short* wkb  = take_u16((size_t)HPn * Dn);
  unsigned short* wvb  = take_u16((size_t)HPn * Dn);
  unsigned short* lwb  = take_u16((size_t)NOn * HPn);
  unsigned short* qbuf = take_u16((size_t)Bn * Hn * Sn * Pn);
  unsigned short* kbuf = take_u16((size_t)Bn * Hn * Sn * Pn);
  unsigned short* vtb  = take_u16((size_t)Bn * Hn * Pn * Sn);   // transposed V
  unsigned short* ctxb = take_u16((size_t)Bn * Sn * HPn);
  float* cmax = take_f32((size_t)Bn * Hn * Sn);
  float* csum = take_f32((size_t)Bn * Hn * Sn);

  convert_f32_bf16<<<512, 256, 0, stream>>>(x,   xb,  Bn * Sn * Dn);
  convert_f32_bf16<<<256, 256, 0, stream>>>(q_w, wqb, HPn * Dn);
  convert_f32_bf16<<<256, 256, 0, stream>>>(k_w, wkb, HPn * Dn);
  convert_f32_bf16<<<256, 256, 0, stream>>>(v_w, wvb, HPn * Dn);
  convert_f32_bf16<<<256, 256, 0, stream>>>(l_w, lwb, NOn * HPn);

  // 2048 waves each, 4 waves / block
  proj_kernel<<<512, 128, 0, stream>>>(xb, wqb, q_b, qbuf, 0);
  proj_kernel<<<512, 128, 0, stream>>>(xb, wkb, k_b, kbuf, 0);
  proj_kernel<<<512, 128, 0, stream>>>(xb, wvb, v_b, vtb, 1);

  colstats_kernel<<<512, 128, 0, stream>>>(qbuf, kbuf, cmax, csum);
  ctx_kernel<<<512, 128, 0, stream>>>(qbuf, kbuf, vtb, cmax, csum, ctxb);
  final_kernel<<<512, 128, 0, stream>>>(ctxb, lwb, l_b, out);
}
